// Attention_23785528886064
// MI455X (gfx1250) — compile-verified, matrix operand order checked by reading, not run on hardware
//
#include <hip/hip_runtime.h>

// Multi-head attention, B=4 N=2048 C=1024 H=16 HD=64, via bf16 WMMA on gfx1250.
// Pipeline: cvt(fp32->bf16, vectorized) -> QKV GEMM (32x32 register-blocked wmma,
// scatter to Q[b,h,n,d], K[b,h,n,d], V^T[b,h,d,n]) -> flash attention (wmma QK^T,
// fp32 online softmax, wmma PV) -> output projection + bias (32x32 wmma, fp32 out).
// Workspace requirement: ~93 MB.

#define BATCH 4
#define SEQ   2048
#define CDIM  1024
#define NH    16
#define HDIM  64
#define QKVD  (3 * CDIM)
#define ATTN_SCALE 0.125f   // HD^-0.5, exact power of two

typedef __bf16 bf16_t;
typedef __bf16 v16bf __attribute__((ext_vector_type(16)));
typedef __bf16 v4bf  __attribute__((ext_vector_type(4)));
typedef float  v8f   __attribute__((ext_vector_type(8)));
typedef float  v4f   __attribute__((ext_vector_type(4)));

// Load one wave-lane's slice of a 16x32 bf16 WMMA A/B fragment.
// Per ISA layout: lane half hf=0 holds K = 0..7 and 16..23; hf=1 holds
// K = 8..15 and 24..31, relative to a contiguous 32-element row at p.
__device__ __forceinline__ v16bf load_frag(const bf16_t* p, int hf) {
    v16bf f;
    const bf16_t* p0 = p + 8 * hf;
    const bf16_t* p1 = p + 16 + 8 * hf;
#pragma unroll
    for (int i = 0; i < 8; ++i) { f[i] = p0[i]; f[8 + i] = p1[i]; }
    return f;
}

__device__ __forceinline__ v8f wmma_bf16(v16bf a, v16bf b, v8f c) {
    return __builtin_amdgcn_wmma_f32_16x16x32_bf16(false, a, false, b,
                                                   (short)0, c, false, false);
}

// Vectorized fp32 -> bf16 conversion: 4 elements per thread per step.
__global__ void cvt_f32_bf16(const float* __restrict__ s, bf16_t* __restrict__ d, int n4) {
    int i = blockIdx.x * blockDim.x + threadIdx.x;
    int stride = gridDim.x * blockDim.x;
    const v4f* sp = (const v4f*)s;
    v4bf* dp = (v4bf*)d;
    for (; i < n4; i += stride) {
        v4f v = sp[i];
        v4bf o;
#pragma unroll
        for (int j = 0; j < 4; ++j) o[j] = (bf16_t)v[j];
        dp[i] = o;
    }
}

// qkv = x @ w_qkv^T.  A = xb [M=B*N, K=C] row-major, B[k][n] = w_qkv[n][k]
// (both contiguous along K).  Each wave computes a 32x32 output tile
// (2 A-frags x 2 B-frags -> 4 wmma per k-step).  Workgroup = 8 waves = 64x128.
// Scatter D into Q/K (b,h,n,d) and V^T (b,h,d,n).
__global__ void qkv_gemm(const bf16_t* __restrict__ xb, const bf16_t* __restrict__ wb,
                         bf16_t* __restrict__ qo, bf16_t* __restrict__ ko,
                         bf16_t* __restrict__ vto) {
    const int wave = threadIdx.x >> 5;
    const int lane = threadIdx.x & 31;
    const int l2 = lane & 15, hf = lane >> 4;
    const int m0 = (blockIdx.x * 2 + (wave >> 2)) * 32;
    const int n0 = (blockIdx.y * 4 + (wave & 3)) * 32;
    const bf16_t* ar0 = xb + (size_t)(m0 + l2) * CDIM;
    const bf16_t* ar1 = xb + (size_t)(m0 + 16 + l2) * CDIM;
    const bf16_t* br0 = wb + (size_t)(n0 + l2) * CDIM;
    const bf16_t* br1 = wb + (size_t)(n0 + 16 + l2) * CDIM;
    v8f acc[2][2] = {{{}, {}}, {{}, {}}};
#pragma unroll 2
    for (int k0 = 0; k0 < CDIM; k0 += 32) {
        v16bf a0 = load_frag(ar0 + k0, hf);
        v16bf a1 = load_frag(ar1 + k0, hf);
        v16bf b0 = load_frag(br0 + k0, hf);
        v16bf b1 = load_frag(br1 + k0, hf);
        acc[0][0] = wmma_bf16(a0, b0, acc[0][0]);
        acc[0][1] = wmma_bf16(a0, b1, acc[0][1]);
        acc[1][0] = wmma_bf16(a1, b0, acc[1][0]);
        acc[1][1] = wmma_bf16(a1, b1, acc[1][1]);
    }
#pragma unroll
    for (int tn = 0; tn < 2; ++tn) {
        const int d = n0 + tn * 16 + l2;
        const int which = d >> 10;          // 0=q 1=k 2=v
        const int cin = d & (CDIM - 1);
        const int h  = cin >> 6;
        const int hd = cin & (HDIM - 1);
#pragma unroll
        for (int tm = 0; tm < 2; ++tm) {
#pragma unroll
            for (int r = 0; r < 8; ++r) {
                const int m    = m0 + tm * 16 + r + 8 * hf;
                const int bidx = m >> 11;   // / SEQ
                const int nrow = m & (SEQ - 1);
                const int bh   = bidx * NH + h;
                bf16_t val = (bf16_t)acc[tm][tn][r];
                if (which == 0)      qo[((size_t)bh * SEQ + nrow) * HDIM + hd] = val;
                else if (which == 1) ko[((size_t)bh * SEQ + nrow) * HDIM + hd] = val;
                else                 vto[((size_t)bh * HDIM + hd) * SEQ + nrow] = val;
            }
        }
    }
}

// Flash attention: grid (SEQ/128, H, B), 8 waves per block, 16 q-rows per wave.
__global__ void attn_flash(const bf16_t* __restrict__ qg, const bf16_t* __restrict__ kg,
                           const bf16_t* __restrict__ vtg, bf16_t* __restrict__ ob) {
    __shared__ bf16_t psh[8][16 * 32];   // per-wave P staging (D-layout -> A-layout)
    const int wave = threadIdx.x >> 5;
    const int lane = threadIdx.x & 31;
    const int l2 = lane & 15, hf = lane >> 4;
    const int b = blockIdx.z, h = blockIdx.y;
    const int bh = b * NH + h;
    const int m0 = blockIdx.x * 128 + wave * 16;

    const bf16_t* qbase = qg  + (size_t)bh * SEQ * HDIM;
    const bf16_t* kbase = kg  + (size_t)bh * SEQ * HDIM;
    const bf16_t* vbase = vtg + (size_t)bh * HDIM * SEQ;   // [HD][N]

    // Q fragments, with softmax scale folded in (0.125 is exact in bf16).
    const bf16_t* qrow = qbase + (size_t)(m0 + l2) * HDIM;
    v16bf aq0 = load_frag(qrow,      hf);
    v16bf aq1 = load_frag(qrow + 32, hf);
#pragma unroll
    for (int i = 0; i < 16; ++i) {
        aq0[i] = (bf16_t)((float)aq0[i] * ATTN_SCALE);
        aq1[i] = (bf16_t)((float)aq1[i] * ATTN_SCALE);
    }

    v8f acc[4] = {{}, {}, {}, {}};
    float mrow[8], lrow[8];
#pragma unroll
    for (int r = 0; r < 8; ++r) { mrow[r] = -1e30f; lrow[r] = 0.f; }

    bf16_t* ps = &psh[wave][0];

    for (int kv0 = 0; kv0 < SEQ; kv0 += 32) {
        // S(16x32) = Qs(16x64) x K^T, two 16x16 D-tiles, K-dim split 2x32.
        v8f s0 = {}, s1 = {};
        const bf16_t* kr0 = kbase + (size_t)(kv0 + l2) * HDIM;
        const bf16_t* kr1 = kbase + (size_t)(kv0 + 16 + l2) * HDIM;
        v16bf b00 = load_frag(kr0,      hf);
        v16bf b01 = load_frag(kr0 + 32, hf);
        v16bf b10 = load_frag(kr1,      hf);
        v16bf b11 = load_frag(kr1 + 32, hf);
        s0 = wmma_bf16(aq0, b00, s0);
        s0 = wmma_bf16(aq1, b01, s0);
        s1 = wmma_bf16(aq0, b10, s1);
        s1 = wmma_bf16(aq1, b11, s1);

        // Online softmax in fp32.  D-layout: lane holds one kv column, VGPR r holds
        // q-row r + 8*hf; row-wise reductions stay inside each 16-lane half.
        float p0[8], p1[8], corr[8];
#pragma unroll
        for (int r = 0; r < 8; ++r) {
            float a = s0[r], c = s1[r];
            float cm = fmaxf(a, c);
#pragma unroll
            for (int msk = 1; msk < 16; msk <<= 1) cm = fmaxf(cm, __shfl_xor(cm, msk, 32));
            float mn = fmaxf(mrow[r], cm);
            corr[r] = __expf(mrow[r] - mn);
            float e0 = __expf(a - mn), e1 = __expf(c - mn);
            float rs = e0 + e1;
#pragma unroll
            for (int msk = 1; msk < 16; msk <<= 1) rs += __shfl_xor(rs, msk, 32);
            lrow[r] = lrow[r] * corr[r] + rs;
            mrow[r] = mn;
            p0[r] = e0; p1[r] = e1;
        }
#pragma unroll
        for (int r = 0; r < 8; ++r) {
            acc[0][r] *= corr[r]; acc[1][r] *= corr[r];
            acc[2][r] *= corr[r]; acc[3][r] *= corr[r];
        }
        // P: D-layout regs -> LDS row-major -> A-layout frag (same-wave DS is in-order).
#pragma unroll
        for (int r = 0; r < 8; ++r) {
            ps[(r + 8 * hf) * 32 + l2]      = (bf16_t)p0[r];
            ps[(r + 8 * hf) * 32 + 16 + l2] = (bf16_t)p1[r];
        }
        v16bf ap = load_frag(ps + l2 * 32, hf);
        // O(16x64) += P(16x32) x V(32x64); V^T storage makes B-frags contiguous.
#pragma unroll
        for (int j = 0; j < 4; ++j) {
            const bf16_t* vr = vbase + (size_t)(16 * j + l2) * SEQ + kv0;
            v16bf bv = load_frag(vr, hf);
            acc[j] = wmma_bf16(ap, bv, acc[j]);
        }
    }
    // Normalize and store attention output as bf16 [B,N,C].
#pragma unroll
    for (int r = 0; r < 8; ++r) {
        float inv = 1.0f / lrow[r];
        int qn = m0 + r + 8 * hf;
        size_t base = ((size_t)b * SEQ + qn) * CDIM + h * HDIM;
#pragma unroll
        for (int j = 0; j < 4; ++j)
            ob[base + j * 16 + l2] = (bf16_t)(acc[j][r] * inv);
    }
}

// out = attn_out @ w_proj^T + b_proj, fp32 output.  32x32 tile per wave.
__global__ void proj_gemm(const bf16_t* __restrict__ ab, const bf16_t* __restrict__ wb,
                          const float* __restrict__ bias, float* __restrict__ out) {
    const int wave = threadIdx.x >> 5;
    const int lane = threadIdx.x & 31;
    const int l2 = lane & 15, hf = lane >> 4;
    const int m0 = (blockIdx.x * 2 + (wave >> 2)) * 32;
    const int n0 = (blockIdx.y * 4 + (wave & 3)) * 32;
    const bf16_t* ar0 = ab + (size_t)(m0 + l2) * CDIM;
    const bf16_t* ar1 = ab + (size_t)(m0 + 16 + l2) * CDIM;
    const bf16_t* br0 = wb + (size_t)(n0 + l2) * CDIM;
    const bf16_t* br1 = wb + (size_t)(n0 + 16 + l2) * CDIM;
    v8f acc[2][2] = {{{}, {}}, {{}, {}}};
#pragma unroll 2
    for (int k0 = 0; k0 < CDIM; k0 += 32) {
        v16bf a0 = load_frag(ar0 + k0, hf);
        v16bf a1 = load_frag(ar1 + k0, hf);
        v16bf b0 = load_frag(br0 + k0, hf);
        v16bf b1 = load_frag(br1 + k0, hf);
        acc[0][0] = wmma_bf16(a0, b0, acc[0][0]);
        acc[0][1] = wmma_bf16(a0, b1, acc[0][1]);
        acc[1][0] = wmma_bf16(a1, b0, acc[1][0]);
        acc[1][1] = wmma_bf16(a1, b1, acc[1][1]);
    }
#pragma unroll
    for (int tn = 0; tn < 2; ++tn) {
        const int n = n0 + tn * 16 + l2;
        const float bv = bias[n];
#pragma unroll
        for (int tm = 0; tm < 2; ++tm) {
#pragma unroll
            for (int r = 0; r < 8; ++r) {
                const int m = m0 + tm * 16 + r + 8 * hf;
                out[(size_t)m * CDIM + n] = acc[tm][tn][r] + bv;
            }
        }
    }
}

extern "C" void kernel_launch(void* const* d_in, const int* in_sizes, int n_in,
                              void* d_out, int out_size, void* d_ws, size_t ws_size,
                              hipStream_t stream) {
    const float* x      = (const float*)d_in[0];
    const float* w_qkv  = (const float*)d_in[1];
    const float* w_proj = (const float*)d_in[2];
    const float* b_proj = (const float*)d_in[3];
    float* out = (float*)d_out;

    char* ws = (char*)d_ws;
    size_t off = 0;
    auto take = [&](size_t elems) {
        bf16_t* p = (bf16_t*)(ws + off);
        off = (off + elems * sizeof(bf16_t) + 255) & ~(size_t)255;
        return p;
    };
    bf16_t* xb     = take((size_t)BATCH * SEQ * CDIM);
    bf16_t* wqkvb  = take((size_t)QKVD * CDIM);
    bf16_t* wprojb = take((size_t)CDIM * CDIM);
    bf16_t* qb     = take((size_t)BATCH * NH * SEQ * HDIM);
    bf16_t* kb     = take((size_t)BATCH * NH * SEQ * HDIM);
    bf16_t* vtb    = take((size_t)BATCH * NH * HDIM * SEQ);
    bf16_t* ob     = take((size_t)BATCH * SEQ * CDIM);

    cvt_f32_bf16<<<2048, 256, 0, stream>>>(x,      xb,     (BATCH * SEQ * CDIM) / 4);
    cvt_f32_bf16<<<1024, 256, 0, stream>>>(w_qkv,  wqkvb,  (QKVD * CDIM) / 4);
    cvt_f32_bf16<<<512,  256, 0, stream>>>(w_proj, wprojb, (CDIM * CDIM) / 4);

    dim3 g2((BATCH * SEQ) / 64, QKVD / 128);
    qkv_gemm<<<g2, 256, 0, stream>>>(xb, wqkvb, qb, kb, vtb);

    dim3 g3(SEQ / 128, NH, BATCH);
    attn_flash<<<g3, 256, 0, stream>>>(qb, kb, vtb, ob);

    dim3 g4((BATCH * SEQ) / 64, CDIM / 128);
    proj_gemm<<<g4, 256, 0, stream>>>(ob, wprojb, b_proj, out);
}
